// CatNet_62955630624912
// MI455X (gfx1250) — matrix operand order repaired and checked by reading, not compiled, not run on hardware
//
#include <hip/hip_runtime.h>
#include <hip/hip_bf16.h>

typedef __attribute__((ext_vector_type(16))) _Float16 v16h;
typedef __attribute__((ext_vector_type(8)))  float    v8f;

#define TT   16
#define HW   784
#define WID  28

union Frag16 { v16h h; uint4 q[2]; };

__device__ __forceinline__ float spike_step(float& u, float inp) {
  u += inp;
  float s = (u >= 1.0f) ? 1.0f : 0.0f;
  u -= s;
  return s;
}

// ---------------------------------------------------------------------------
// Weight conversion f32 -> f16 (natural flatten is already GEMM-K order).
// ---------------------------------------------------------------------------
__global__ void k_cvt(const float* __restrict__ w2, const float* __restrict__ w3,
                      const float* __restrict__ wf1,
                      _Float16* __restrict__ W2h, _Float16* __restrict__ W3h,
                      _Float16* __restrict__ WF1h) {
  const int n2 = 64 * 32 * 9, n3 = 64 * 64 * 9, nf = 128 * 12544;
  int gid = blockIdx.x * blockDim.x + threadIdx.x;
  if (gid < n2)                { W2h[gid] = (_Float16)w2[gid]; return; }
  gid -= n2;
  if (gid < n3)                { W3h[gid] = (_Float16)w3[gid]; return; }
  gid -= n3;
  if (gid < nf)                { WF1h[gid] = (_Float16)wf1[gid]; }
}

__global__ void k_zero(float* __restrict__ p, int n) {
  int gid = blockIdx.x * blockDim.x + threadIdx.x;
  if (gid < n) p[gid] = 0.0f;
}

// ---------------------------------------------------------------------------
// Stage 1: conv1_ (64ch) and conv1 (32ch), 1 input channel, + IF spike scan.
// ---------------------------------------------------------------------------
__global__ void k_conv1_spike(const float* __restrict__ x,
                              const float* __restrict__ w1a, const float* __restrict__ b1a,
                              const float* __restrict__ w1,  const float* __restrict__ b1,
                              _Float16* __restrict__ X1,   // [n][t][64][784]
                              _Float16* __restrict__ H0) { // [n][t][32][784]
  int gid = blockIdx.x * blockDim.x + threadIdx.x;
  if (gid >= 64 * 96 * HW) return;
  const int pix  = gid % HW;
  const int oc96 = (gid / HW) % 96;
  const int n    = gid / (HW * 96);
  const int h = pix / WID, w = pix % WID;
  const bool isA = (oc96 < 64);
  const int  oc  = isA ? oc96 : (oc96 - 64);
  const float* wk = isA ? (w1a + oc * 9) : (w1 + oc * 9);
  const float  bs = isA ? b1a[oc] : b1[oc];
  float wv[9];
#pragma unroll
  for (int i = 0; i < 9; ++i) wv[i] = wk[i];
  float u = 0.0f;
  for (int t = 0; t < TT; ++t) {
    float acc = bs;
#pragma unroll
    for (int r = 0; r < 3; ++r)
#pragma unroll
      for (int c = 0; c < 3; ++c) {
        int ih = h + r - 1, iw = w + c - 1;
        bool inb = (ih >= 0) & (ih < WID) & (iw >= 0) & (iw < WID);
        int ih2 = inb ? ih : 0, iw2 = inb ? iw : 0;
        float v = x[(n * HW + ih2 * WID + iw2) * TT + t];
        acc += inb ? wv[r * 3 + c] * v : 0.0f;
      }
    float s = spike_step(u, acc);
    if (isA) X1[((size_t)(n * TT + t) * 64 + oc) * HW + pix] = (_Float16)s;
    else     H0[((size_t)(n * TT + t) * 32 + oc) * HW + pix] = (_Float16)s;
  }
}

// ---------------------------------------------------------------------------
// WMMA implicit-GEMM 3x3 conv + optional skip + IF spike scan.
// Block = 128 threads = 4 waves; wave w owns oc-tile w. Weight (A) fragments
// are preloaded into registers ONCE (reused across all 16 timesteps), so the
// steady-state inner loop is 2x ds_load_b128 + 1 wmma per chunk. The K x 16
// im2col panel is rebuilt in LDS per timestep; each build thread owns one
// column, with halo offsets/masks precomputed outside the t loop.
// ---------------------------------------------------------------------------
template <int IC, bool SKIP>
__global__ void __launch_bounds__(128, 1)
k_conv_wmma_spike(const _Float16* __restrict__ W,    // [64][IC*9]
                  const float* __restrict__ bias,    // [64]
                  const _Float16* __restrict__ Src,  // [n][t][IC][784]
                  const _Float16* __restrict__ Skip, // [n][t][64][784]
                  _Float16* __restrict__ Dst) {      // [n][t][64][784]
  constexpr int K      = IC * 9;
  constexpr int NCHUNK = K / 32;
  constexpr int KPAD   = K + 8;
  __shared__ _Float16 Bs[16 * KPAD];

  const int tid    = threadIdx.x;
  const int lane   = tid & 31;
  const int octile = tid >> 5;           // wave id = oc tile
  const int half   = lane >> 4;
  const int lm     = lane & 15;
  const int ptile  = blockIdx.x;         // 49 tiles of 16 pixels
  const int n      = blockIdx.y;
  const int pix = ptile * 16 + lm;       // B/C column (N dim)
  const int ocA = octile * 16 + lm;      // A row (M dim)

  // ---- im2col build geometry: thread owns column bcol, ic stride 8 ----
  const int bcol = tid >> 3;             // 0..15
  const int bsub = tid & 7;              // ic offset
  int  off9[9];
  bool inb9[9];
  {
    int bp = ptile * 16 + bcol;
    int ph = bp / WID, pw = bp % WID;
#pragma unroll
    for (int rem = 0; rem < 9; ++rem) {
      int kr = rem / 3, kc = rem % 3;
      int ih = ph + kr - 1, iw = pw + kc - 1;
      bool inb = (ih >= 0) & (ih < WID) & (iw >= 0) & (iw < WID);
      inb9[rem] = inb;
      off9[rem] = (inb ? ih : 0) * WID + (inb ? iw : 0);
    }
  }

  // ---- preload A fragments (weights): registers, reused for all t ----
  Frag16 aFr[NCHUNK];
  {
    const _Float16* wRow = W + (size_t)ocA * K;
#pragma unroll
    for (int ch = 0; ch < NCHUNK; ++ch) {
      aFr[ch].q[0] = *(const uint4*)(wRow + ch * 32 + half * 8);
      aFr[ch].q[1] = *(const uint4*)(wRow + ch * 32 + half * 8 + 16);
    }
  }

  float u[8];
#pragma unroll
  for (int r = 0; r < 8; ++r) u[r] = 0.0f;

  for (int t = 0; t < TT; ++t) {
    const _Float16* srcT = Src + (size_t)(n * TT + t) * IC * HW;
    // ---- cooperative im2col panel build (no div/mod in hot loop) ----
    for (int ic = bsub; ic < IC; ic += 8) {
      const _Float16* s = srcT + (size_t)ic * HW;
      _Float16* d = Bs + bcol * KPAD + ic * 9;
#pragma unroll
      for (int rem = 0; rem < 9; ++rem)
        d[rem] = inb9[rem] ? s[off9[rem]] : (_Float16)0.0f;
    }
    if (t + 1 < TT)  // hint next timestep's panel toward the caches
      __builtin_prefetch((const void*)(srcT + (size_t)IC * HW + tid * 32), 0, 1);
    __syncthreads();

    // ---- WMMA: 2 ds_load_b128 + 1 wmma per chunk (A already resident) ----
    v8f acc = {};
    const _Float16* bRow = Bs + lm * KPAD + half * 8;
#pragma unroll
    for (int ch = 0; ch < NCHUNK; ++ch) {
      Frag16 b;
      b.q[0] = *(const uint4*)(bRow + ch * 32);
      b.q[1] = *(const uint4*)(bRow + ch * 32 + 16);
      acc = __builtin_amdgcn_wmma_f32_16x16x32_f16(
          false, aFr[ch].h, false, b.h, (short)0, acc, false, false);
    }

    // ---- IF neuron update + spike store (C: VGPR r -> M = r + 8*half) ----
#pragma unroll
    for (int r = 0; r < 8; ++r) {
      int oce = octile * 16 + r + 8 * half;
      float pre = acc[r] + bias[oce];
      if (SKIP) pre += (float)Skip[((size_t)(n * TT + t) * 64 + oce) * HW + pix];
      float s = spike_step(u[r], pre);
      Dst[((size_t)(n * TT + t) * 64 + oce) * HW + pix] = (_Float16)s;
    }
    __syncthreads();   // all waves done with Bs before next t overwrites it
  }
}

// ---------------------------------------------------------------------------
// 2x2 sum-pool * 1.1 + IF spike. H2 [n][t][64][784] -> H3 [n][t][64][196]
// ---------------------------------------------------------------------------
__global__ void k_pool_spike(const _Float16* __restrict__ H2,
                             _Float16* __restrict__ H3) {
  int gid = blockIdx.x * blockDim.x + threadIdx.x;
  if (gid >= 64 * 64 * 196) return;
  const int p = gid % 196;
  const int c = (gid / 196) % 64;
  const int n = gid / (196 * 64);
  const int y = p / 14, xx = p % 14;
  float u = 0.0f;
  for (int t = 0; t < TT; ++t) {
    const _Float16* src = H2 + ((size_t)(n * TT + t) * 64 + c) * HW;
    float s4 = (float)src[(2 * y) * WID + 2 * xx]     + (float)src[(2 * y) * WID + 2 * xx + 1]
             + (float)src[(2 * y + 1) * WID + 2 * xx] + (float)src[(2 * y + 1) * WID + 2 * xx + 1];
    float s = spike_step(u, 1.1f * s4);
    H3[((size_t)(n * TT + t) * 64 + c) * 196 + p] = (_Float16)s;
  }
}

// ---------------------------------------------------------------------------
// fc1 GEMM: M=128(oc) x N=64(batch) x K=12544. Block = 256 threads = 8 waves
// (one per oc tile); K split over 14 grid slices of 896. The flattened K is
// contiguous in H3, so the B panel is staged into LDS with
// global_load_async_to_lds_b128 (ASYNCcnt / s_wait_asynccnt), then consumed
// as ds_load_b128 pairs. Partials accumulate via global f32 atomics.
// ---------------------------------------------------------------------------
__global__ void __launch_bounds__(256)
k_fc1_wmma(const _Float16* __restrict__ Wf,  // [128][12544]
           const _Float16* __restrict__ H3,  // [n][t][12544]  (c*196+p flat)
           float* __restrict__ PRE) {        // [n][128][16]
  constexpr int KTOT  = 12544;
  constexpr int KSL   = 896;                 // 28 chunks of 32
  constexpr int KPAD  = KSL + 8;
  __shared__ _Float16 Bs[16 * KPAD];

  const int tid    = threadIdx.x;
  const int lane   = tid & 31;
  const int octile = tid >> 5;               // 0..7
  const int half   = lane >> 4;
  const int lm     = lane & 15;
  const int ntile  = blockIdx.x;             // 0..3 (batch tiles)
  const int c0     = blockIdx.y * 28;        // chunk base (14 slices)
  const int ncol = ntile * 16 + lm;
  const int ocA  = octile * 16 + lm;

  for (int t = 0; t < TT; ++t) {
    // ---- async bulk copy of B panel: 16 cols x 1792B, 16B per lane-op ----
    for (int uu = tid; uu < 16 * 112; uu += 256) {   // 1792/256 = 7 (uniform)
      int col = uu / 112;
      int ob  = (uu - col * 112) * 16;               // byte offset in column
      const char* g = (const char*)(H3 + (size_t)((ntile * 16 + col) * TT + t) * KTOT
                                       + (size_t)c0 * 32) + ob;
      unsigned      ldsb = (unsigned)(uintptr_t)((const char*)&Bs[col * KPAD]) + (unsigned)ob;
      unsigned long long ga = (unsigned long long)(uintptr_t)g;
      asm volatile("global_load_async_to_lds_b128 %0, %1, off"
                   :: "v"(ldsb), "v"(ga) : "memory");
    }
    asm volatile("s_wait_asynccnt 0x0" ::: "memory");
    __syncthreads();

    // ---- WMMA over the 28-chunk slice ----
    v8f acc = {};
    const _Float16* wRow = Wf + (size_t)ocA * KTOT + (size_t)c0 * 32;
    const _Float16* bRow = Bs + lm * KPAD + half * 8;
    for (int ch = 0; ch < 28; ++ch) {
      Frag16 a, b;
      a.q[0] = *(const uint4*)(wRow + ch * 32 + half * 8);
      a.q[1] = *(const uint4*)(wRow + ch * 32 + half * 8 + 16);
      b.q[0] = *(const uint4*)(bRow + ch * 32);
      b.q[1] = *(const uint4*)(bRow + ch * 32 + 16);
      acc = __builtin_amdgcn_wmma_f32_16x16x32_f16(
          false, a.h, false, b.h, (short)0, acc, false, false);
    }
#pragma unroll
    for (int r = 0; r < 8; ++r) {
      int oce = octile * 16 + r + 8 * half;
      atomicAdd(&PRE[((size_t)ncol * 128 + oce) * TT + t], acc[r]);
    }
    __syncthreads();
  }
}

__global__ void k_fc1_spike(const float* __restrict__ PRE, const float* __restrict__ bf1,
                            _Float16* __restrict__ F1) {  // [n][128][16]
  int gid = blockIdx.x * blockDim.x + threadIdx.x;
  if (gid >= 64 * 128) return;
  const int c = gid % 128;
  float u = 0.0f;
  const float bb = bf1[c];
  for (int t = 0; t < TT; ++t) {
    float s = spike_step(u, PRE[(size_t)gid * TT + t] + bb);
    F1[(size_t)gid * TT + t] = (_Float16)s;
  }
}

__global__ void k_fc2(const _Float16* __restrict__ F1, const float* __restrict__ wf2,
                      const float* __restrict__ bf2, float* __restrict__ out) {
  int gid = blockIdx.x * blockDim.x + threadIdx.x;
  if (gid >= 64 * 10) return;
  const int o = gid % 10, n = gid / 10;
  float acc = 0.0f;
  for (int t = 0; t < TT; ++t) {
    float tmp = bf2[o];
    for (int c = 0; c < 128; ++c)
      tmp += wf2[o * 128 + c] * (float)F1[((size_t)n * 128 + c) * TT + t];
    acc += tmp;
  }
  out[gid] = acc * (1.0f / (float)TT);
}

// ---------------------------------------------------------------------------
extern "C" void kernel_launch(void* const* d_in, const int* in_sizes, int n_in,
                              void* d_out, int out_size, void* d_ws, size_t ws_size,
                              hipStream_t stream) {
  const float* x   = (const float*)d_in[0];
  const float* w1a = (const float*)d_in[1];
  const float* b1a = (const float*)d_in[2];
  const float* w1  = (const float*)d_in[3];
  const float* b1  = (const float*)d_in[4];
  const float* w2  = (const float*)d_in[5];
  const float* b2  = (const float*)d_in[6];
  const float* w3  = (const float*)d_in[7];
  const float* b3  = (const float*)d_in[8];
  const float* wf1 = (const float*)d_in[9];
  const float* bf1 = (const float*)d_in[10];
  const float* wf2 = (const float*)d_in[11];
  const float* bf2 = (const float*)d_in[12];
  float* out = (float*)d_out;

  // Workspace layout (bytes, 256-aligned). H2 aliases X1; H3/PRE/F1 alias H0.
  char* ws = (char*)d_ws;
  _Float16* W2h  = (_Float16*)(ws + 0);                        //   36,864 B
  _Float16* W3h  = (_Float16*)(ws + 36864);                    //   73,728 B
  _Float16* WF1h = (_Float16*)(ws + 110592);                   // 3,211,264 B
  _Float16* X1   = (_Float16*)(ws + 3321856);                  // 102,760,448 B
  _Float16* H0   = (_Float16*)(ws + 106082304);                //  51,380,224 B
  _Float16* H1   = (_Float16*)(ws + 157462528);                // 102,760,448 B
  _Float16* H2   = X1;                                         // alias (X1 dead)
  _Float16* H3   = H0;                                         // alias (H0 dead)
  float*    PRE  = (float*)(ws + 106082304 + 25690112);        //     524,288 B
  _Float16* F1   = (_Float16*)(ws + 106082304 + 25690112 + 524288);

  // 0) weight conversion + zero fc1 pre-activation accumulator
  {
    int n = 64 * 32 * 9 + 64 * 64 * 9 + 128 * 12544;
    k_cvt<<<(n + 255) / 256, 256, 0, stream>>>(w2, w3, wf1, W2h, W3h, WF1h);
    k_zero<<<(64 * 128 * TT + 255) / 256, 256, 0, stream>>>(PRE, 64 * 128 * TT);
  }
  // 1) conv1_ / conv1 + spike
  {
    int n = 64 * 96 * HW;
    k_conv1_spike<<<(n + 255) / 256, 256, 0, stream>>>(x, w1a, b1a, w1, b1, X1, H0);
  }
  // 2) conv2 + x1 skip + spike  (WMMA, IC=32, 4 waves share LDS panel)
  k_conv_wmma_spike<32, true><<<dim3(49, 64), 128, 0, stream>>>(W2h, b2, H0, X1, H1);
  // 3) conv3 + spike            (WMMA, IC=64)
  k_conv_wmma_spike<64, false><<<dim3(49, 64), 128, 0, stream>>>(W3h, b3, H1, nullptr, H2);
  // 4) pool + spike
  k_pool_spike<<<(64 * 64 * 196 + 255) / 256, 256, 0, stream>>>(H2, H3);
  // 5) fc1 (WMMA, async-LDS staging, K-split + f32 atomics) then spike scan
  k_fc1_wmma<<<dim3(4, 14), 256, 0, stream>>>(WF1h, H3, PRE);
  k_fc1_spike<<<(64 * 128 + 255) / 256, 256, 0, stream>>>(PRE, bf1, F1);
  // 6) fc2 + temporal mean
  k_fc2<<<(640 + 255) / 256, 256, 0, stream>>>(F1, wf2, bf2, out);
}
